// LinearDeepSeekV3_6330781794367
// MI455X (gfx1250) — 8-bit matrix operand order repaired and checked by reading, compiled, not run on hardware
//
#include <hip/hip_runtime.h>
#include <hip/hip_bf16.h>

typedef __attribute__((ext_vector_type(16))) int   v16i;
typedef __attribute__((ext_vector_type(8)))  float v8f;

#define K_DIM   7168
#define N_DIM   512
#define KB_CNT  56
#define BLK     128
#define M_TILE  32
#define ROW_STRIDE 144

__device__ __forceinline__ unsigned int f32_to_e4m3_sw(float x) {
    unsigned int sign = (__float_as_uint(x) >> 31) << 7;
    float ax = fminf(fabsf(x), 448.0f);
    unsigned int u = __float_as_uint(ax);
    int e = (int)((u >> 23) & 0xff) - 127;
    unsigned int q;
    if (e >= -6) {
        unsigned int m = u & 0x7fffff;
        q = ((unsigned int)(e + 7) << 3) | (m >> 20);
        unsigned int rem = m & 0xfffff;
        if (rem > 0x80000u || (rem == 0x80000u && (q & 1u))) q++;
        if (q > 0x7eu) q = 0x7eu;
    } else {
        q = (unsigned int)__builtin_rintf(ax * 512.0f);
    }
    return sign | q;
}

__device__ __forceinline__ unsigned int pack4_e4m3(float a, float b, float c, float d) {
#if __has_builtin(__builtin_amdgcn_cvt_pk_fp8_f32)
    int lo  = __builtin_amdgcn_cvt_pk_fp8_f32(a, b, 0,  false);
    int all = __builtin_amdgcn_cvt_pk_fp8_f32(c, d, lo, true);
    return (unsigned int)all;
#else
    return f32_to_e4m3_sw(a) | (f32_to_e4m3_sw(b) << 8)
         | (f32_to_e4m3_sw(c) << 16) | (f32_to_e4m3_sw(d) << 24);
#endif
}

__global__ void __launch_bounds__(256) wcvt_kernel(const float* __restrict__ wq,
                                                   unsigned int* __restrict__ w8,
                                                   int n4) {
    int i = blockIdx.x * blockDim.x + threadIdx.x;
    if (i >= n4) return;
    float4 f = ((const float4*)wq)[i];
    w8[i] = pack4_e4m3(f.x, f.y, f.z, f.w);
}

__global__ void __launch_bounds__(256) fp8_gemm_kernel(const float* __restrict__ x,
                                                       const unsigned char* __restrict__ w8,
                                                       const float* __restrict__ ws,
                                                       float* __restrict__ y) {
    __shared__ __align__(16) unsigned char Ab[M_TILE * ROW_STRIDE];
    __shared__ float As[M_TILE];

    const int tid  = threadIdx.x;
    const int lane = tid & 31;
    const int wave = tid >> 5;
    const int mWG  = blockIdx.x * M_TILE;

    const int qrow = wave * 4 + (lane >> 3);
    const int qcol = (lane & 7) * 16;
    const float* xrow = x + (size_t)(mWG + qrow) * K_DIM + qcol;

    const int h  = lane >> 4;
    const int ln = lane & 15;
    const int nBase = wave * 64;

    const int wsBase = __builtin_amdgcn_readfirstlane((nBase >> 7) * KB_CNT);

    v8f acc[2][4];
#pragma unroll
    for (int mt = 0; mt < 2; ++mt)
#pragma unroll
        for (int nt = 0; nt < 4; ++nt)
            acc[mt][nt] = (v8f){};

    float4 stage[4];
    {
        const float4* xv = (const float4*)(xrow);
#pragma unroll
        for (int i = 0; i < 4; ++i) stage[i] = xv[i];
    }

    for (int kb = 0; kb < KB_CNT; ++kb) {
        __syncthreads();

        float amax = 0.0f;
#pragma unroll
        for (int i = 0; i < 4; ++i)
            amax = fmaxf(amax, fmaxf(fmaxf(fabsf(stage[i].x), fabsf(stage[i].y)),
                                     fmaxf(fabsf(stage[i].z), fabsf(stage[i].w))));
        amax = fmaxf(amax, __shfl_xor(amax, 1, 32));
        amax = fmaxf(amax, __shfl_xor(amax, 2, 32));
        amax = fmaxf(amax, __shfl_xor(amax, 4, 32));
        amax = fmaxf(amax, 1e-20f);
        float s = amax * (1.0f / 448.0f);
        float r = 448.0f / amax;

        unsigned int q[4];
#pragma unroll
        for (int i = 0; i < 4; ++i)
            q[i] = pack4_e4m3(stage[i].x * r, stage[i].y * r,
                              stage[i].z * r, stage[i].w * r);
        *(uint4*)(Ab + qrow * ROW_STRIDE + qcol) = make_uint4(q[0], q[1], q[2], q[3]);
        if ((lane & 7) == 0) As[qrow] = s;

        if (kb + 1 < KB_CNT) {
            const float4* xv = (const float4*)(xrow + (size_t)(kb + 1) * BLK);
#pragma unroll
            for (int i = 0; i < 4; ++i) stage[i] = xv[i];
            if (kb + 2 < KB_CNT)
                __builtin_prefetch(xrow + (size_t)(kb + 2) * BLK, 0, 0);
        }

        __syncthreads();

        float bs = ws[wsBase + kb];

        v16i afrag[2];
#pragma unroll
        for (int mt = 0; mt < 2; ++mt) {
            const unsigned char* ab = Ab + (mt * 16 + ln) * ROW_STRIDE + 16 * h;
            v16i a;
#pragma unroll
            for (int q2 = 0; q2 < 8; ++q2) {
                unsigned long long d = *(const unsigned long long*)(ab + 32 * (q2 >> 1) + 8 * (q2 & 1));
                a[2 * q2]     = (int)(unsigned int)d;
                a[2 * q2 + 1] = (int)(unsigned int)(d >> 32);
            }
            afrag[mt] = a;
        }

        float scb[2][8];
#pragma unroll
        for (int mt = 0; mt < 2; ++mt) {
            const float* sp = &As[mt * 16 + 8 * h];
#pragma unroll
            for (int v = 0; v < 8; ++v) scb[mt][v] = sp[v] * bs;
        }

        const unsigned char* wb = w8 + (size_t)(nBase + ln) * K_DIM + (size_t)kb * BLK + h * 16;
#pragma unroll
        for (int nt = 0; nt < 4; ++nt) {
            const unsigned char* wp = wb + (size_t)nt * 16 * K_DIM;
            uint4 g0 = *(const uint4*)(wp);
            uint4 g1 = *(const uint4*)(wp + 32);
            uint4 g2 = *(const uint4*)(wp + 64);
            uint4 g3 = *(const uint4*)(wp + 96);
            v16i b;
            b[0]  = (int)g0.x; b[1]  = (int)g0.y; b[2]  = (int)g0.z; b[3]  = (int)g0.w;
            b[4]  = (int)g1.x; b[5]  = (int)g1.y; b[6]  = (int)g1.z; b[7]  = (int)g1.w;
            b[8]  = (int)g2.x; b[9]  = (int)g2.y; b[10] = (int)g2.z; b[11] = (int)g2.w;
            b[12] = (int)g3.x; b[13] = (int)g3.y; b[14] = (int)g3.z; b[15] = (int)g3.w;
#pragma unroll
            for (int mt = 0; mt < 2; ++mt) {
                v8f z = {};
                v8f p = __builtin_amdgcn_wmma_f32_16x16x128_fp8_fp8(
                            afrag[mt], b, (short)0, z, false, false);
#pragma unroll
                for (int v = 0; v < 8; ++v)
                    acc[mt][nt][v] += p[v] * scb[mt][v];
            }
        }
    }

#pragma unroll
    for (int mt = 0; mt < 2; ++mt) {
        int rbase = mWG + mt * 16 + 8 * h;
#pragma unroll
        for (int nt = 0; nt < 4; ++nt) {
            int col = nBase + nt * 16 + ln;
#pragma unroll
            for (int v = 0; v < 8; ++v)
                y[(size_t)(rbase + v) * N_DIM + col] = acc[mt][nt][v];
        }
    }
}

extern "C" void kernel_launch(void* const* d_in, const int* in_sizes, int n_in,
                              void* d_out, int out_size, void* d_ws, size_t ws_size,
                              hipStream_t stream) {
    const float* x  = (const float*)d_in[0];
    const float* wq = (const float*)d_in[1];
    const float* ws = (const float*)d_in[2];

    const int M = in_sizes[0] / K_DIM;

    unsigned int* w8 = (unsigned int*)d_ws;

    int n4 = (N_DIM * K_DIM) / 4;
    wcvt_kernel<<<(n4 + 255) / 256, 256, 0, stream>>>(wq, w8, n4);

    fp8_gemm_kernel<<<M / M_TILE, 256, 0, stream>>>(
        x, (const unsigned char*)w8, ws, (float*)d_out);
}
